// MultiHeadAttention_14353780703668
// MI455X (gfx1250) — matrix operand order repaired and checked by reading, not compiled, not run on hardware
//
#include <hip/hip_runtime.h>
#include <hip/hip_bf16.h>

// MHA forward, MI455X (gfx1250, wave32, WMMA f16->f32 path).
#define IN_CH  1024
#define N_HEAD 16
#define K_CH   64
#define SEQ    2048
#define BATCH  2
#define ROWS   (BATCH * SEQ)   // 4096 total query rows

typedef __attribute__((ext_vector_type(16))) _Float16 v16h;
typedef __attribute__((ext_vector_type(8)))  float    v8f;

#define WMMA_F16(a, b, c) \
  __builtin_amdgcn_wmma_f32_16x16x32_f16(false, (a), false, (b), (short)0, (c), false, false)

// ---------------------------------------------------------------------------
// Fragment loaders per CDNA5 ISA §7.12.2 (wave32 layouts), all sources f16
// with the K-pair contiguous in memory (merges into b128 loads).
//   lane = m + 16*h  (m = row-of-A / col-of-B, h = lane half)
// A 16x32: VGPR v<4 -> K = 2v + 8h ; v>=4 -> K = 16 + 2(v-4) + 8h
// B 32x16: VGPR v<4 -> K = 2v + 8h ; v>=4 -> K = 16 + 2(v-4) + 8h ; element (k,n) read from src[n*ld + k]
// C/D 16x16 f32: VGPR r -> row r + 8h, lane%16 = N
// ---------------------------------------------------------------------------

static __device__ __forceinline__ v16h
load_a_f16(const _Float16* __restrict__ base, int ld, int m, int h, int kbase) {
  v16h a{};
#pragma unroll
  for (int v = 0; v < 8; ++v) {
    const int koff = (v < 4) ? (2 * v + 8 * h) : (16 + 2 * (v - 4) + 8 * h);
    const _Float16* p = base + (size_t)m * ld + kbase + koff;
    a[2 * v]     = p[0];
    a[2 * v + 1] = p[1];
  }
  return a;
}

// B element (k, n) = src[n*ld + k]   (column n is a contiguous K-run)
static __device__ __forceinline__ v16h
load_bT_f16(const _Float16* __restrict__ src, int ld, int n, int h, int kbase) {
  v16h b{};
#pragma unroll
  for (int v = 0; v < 8; ++v) {
    const int k = kbase + ((v < 4) ? (2 * v + 8 * h) : (16 + 2 * (v - 4) + 8 * h));
    const _Float16* p = src + (size_t)n * ld + k;
    b[2 * v]     = p[0];
    b[2 * v + 1] = p[1];
  }
  return b;
}

// ---------------------------------------------------------------------------
// Pass 0a: elementwise fp32 -> f16 (activations q/k/v). Streaming.
// ---------------------------------------------------------------------------
__global__ void __launch_bounds__(256)
cvt_f16(const float* __restrict__ src, _Float16* __restrict__ dst, int n) {
  const int i = blockIdx.x * 256 + threadIdx.x;
  if (i < n) dst[i] = (_Float16)src[i];
}

// ---------------------------------------------------------------------------
// Pass 0b: W [K=1024][N=1024] fp32 -> Wt [N][K] f16 (tiled LDS transpose,
// coalesced on both sides).
// ---------------------------------------------------------------------------
__global__ void __launch_bounds__(256)
wcvt_t(const float* __restrict__ W, _Float16* __restrict__ Wt) {
  __shared__ float tile[32][33];
  const int tx = threadIdx.x, ty = threadIdx.y;     // block (32, 8)
  const int c0 = blockIdx.x * 32, r0 = blockIdx.y * 32;
#pragma unroll
  for (int yy = ty; yy < 32; yy += 8)
    tile[yy][tx] = W[(size_t)(r0 + yy) * IN_CH + c0 + tx];
  __syncthreads();
#pragma unroll
  for (int yy = ty; yy < 32; yy += 8)
    Wt[(size_t)(c0 + yy) * IN_CH + r0 + tx] = (_Float16)tile[tx][yy];
}

// ---------------------------------------------------------------------------
// Pass 1: projection  out = head_split(X @ W + bias)
// One wave computes a 16x64 output block (4 WMMA accumulators). Two-phase
// loop (even/odd fragment register sets, no rotation copies): while one set's
// loads are in flight the other set's WMMAs issue, so waits are partial.
// vmode=0: out [b][h][s][64]; vmode=1 (V): out [b][h][64][2048].
// ---------------------------------------------------------------------------
__global__ void __launch_bounds__(32)
proj16(const _Float16* __restrict__ Xf, const _Float16* __restrict__ Wt,
       const float* __restrict__ bias, _Float16* __restrict__ out, int vmode) {
  const int lane = threadIdx.x & 31;
  const int m = lane & 15, h = lane >> 4;
  const int col0 = blockIdx.x * 64;   // 16 blocks over 1024 out channels
  const int row0 = blockIdx.y * 16;   // 256 blocks over 4096 rows
  const _Float16* Arow = Xf + (size_t)row0 * IN_CH;

  v8f acc0 = {}, acc1 = {}, acc2 = {}, acc3 = {};

  // set 0 for kb, set 1 for kb+32
  v16h a0  = load_a_f16(Arow, IN_CH, m, h, 0);
  v16h b00 = load_bT_f16(Wt, IN_CH, col0 +  0 + m, h, 0);
  v16h b01 = load_bT_f16(Wt, IN_CH, col0 + 16 + m, h, 0);
  v16h b02 = load_bT_f16(Wt, IN_CH, col0 + 32 + m, h, 0);
  v16h b03 = load_bT_f16(Wt, IN_CH, col0 + 48 + m, h, 0);

#pragma unroll 1
  for (int kb = 0; kb < IN_CH; kb += 64) {
    // phase 0: issue loads for kb+32, consume kb
    v16h a1  = load_a_f16(Arow, IN_CH, m, h, kb + 32);
    v16h b10 = load_bT_f16(Wt, IN_CH, col0 +  0 + m, h, kb + 32);
    v16h b11 = load_bT_f16(Wt, IN_CH, col0 + 16 + m, h, kb + 32);
    v16h b12 = load_bT_f16(Wt, IN_CH, col0 + 32 + m, h, kb + 32);
    v16h b13 = load_bT_f16(Wt, IN_CH, col0 + 48 + m, h, kb + 32);
    acc0 = WMMA_F16(a0, b00, acc0);
    acc1 = WMMA_F16(a0, b01, acc1);
    acc2 = WMMA_F16(a0, b02, acc2);
    acc3 = WMMA_F16(a0, b03, acc3);
    // phase 1: issue loads for kb+64, consume kb+32
    if (kb + 64 < IN_CH) {
      a0  = load_a_f16(Arow, IN_CH, m, h, kb + 64);
      b00 = load_bT_f16(Wt, IN_CH, col0 +  0 + m, h, kb + 64);
      b01 = load_bT_f16(Wt, IN_CH, col0 + 16 + m, h, kb + 64);
      b02 = load_bT_f16(Wt, IN_CH, col0 + 32 + m, h, kb + 64);
      b03 = load_bT_f16(Wt, IN_CH, col0 + 48 + m, h, kb + 64);
    }
    acc0 = WMMA_F16(a1, b10, acc0);
    acc1 = WMMA_F16(a1, b11, acc1);
    acc2 = WMMA_F16(a1, b12, acc2);
    acc3 = WMMA_F16(a1, b13, acc3);
  }

  v8f accs[4] = {acc0, acc1, acc2, acc3};
#pragma unroll
  for (int j = 0; j < 4; ++j) {
    const int col = col0 + j * 16 + m;
    const int hd = col >> 6, d = col & 63;
    const float bv = bias[col];
#pragma unroll
    for (int r = 0; r < 8; ++r) {
      const int gr = row0 + r + 8 * h;
      const int bb = gr >> 11, s = gr & (SEQ - 1);
      const _Float16 val = (_Float16)(accs[j][r] + bv);
      if (vmode == 0)
        out[(((size_t)bb * N_HEAD + hd) * SEQ + s) * K_CH + d] = val;
      else
        out[(((size_t)bb * N_HEAD + hd) * K_CH + d) * SEQ + s] = val;
    }
  }
}

// ---------------------------------------------------------------------------
// Pass 2: attention. One workgroup (4 waves) owns 16 query rows of one
// (b,h): full score stripe S[16][2048] in LDS (f32 exp-values), softmax
// stats, single non-temporal fp32 attn write (the 537MB roofline limiter,
// normalization fused into the streaming store), then P@V from LDS f16
// probs against transposed-V. LDS ~192.6KB of the 320KB WGP pool.
// ---------------------------------------------------------------------------
__global__ void __launch_bounds__(128)
attn16(const _Float16* __restrict__ Qh, const _Float16* __restrict__ Kh,
       const _Float16* __restrict__ Vt, float* __restrict__ attn,
       _Float16* __restrict__ O) {
  extern __shared__ char smem_raw[];
  float*    Ssc    = (float*)smem_raw;                                 // [16][SEQ] f32
  _Float16* Pp     = (_Float16*)(smem_raw + 16 * SEQ * 4);             // [16][SEQ] f16
  float*    red    = (float*)(smem_raw + 16 * SEQ * 4 + 16 * SEQ * 2); // [16][8]
  float*    invRow = red + 16 * 8;                                     // [16]

  const int t = threadIdx.x;
  const int lane = t & 31, wave = t >> 5;
  const int m = lane & 15, h = lane >> 4;
  const int bh = blockIdx.x >> 7;          // b*N_HEAD + head
  const int q0 = (blockIdx.x & 127) * 16;  // query tile base

  const _Float16* Qbh = Qh + (size_t)bh * SEQ * K_CH;
  const _Float16* Kbh = Kh + (size_t)bh * SEQ * K_CH;
  const _Float16* Vbh = Vt + (size_t)bh * K_CH * SEQ;   // [64][2048]

  // Q tile fragments (d = 64 -> two K=32 chunks), reused by all key tiles.
  const v16h aq0 = load_a_f16(Qbh + (size_t)q0 * K_CH, K_CH, m, h, 0);
  const v16h aq1 = load_a_f16(Qbh + (size_t)q0 * K_CH, K_CH, m, h, 32);

  // ---- scores: each wave handles every 4th key tile, two tiles/iteration --
  {
    v16h b00 = load_bT_f16(Kbh, K_CH, wave * 16 + m, h, 0);
    v16h b01 = load_bT_f16(Kbh, K_CH, wave * 16 + m, h, 32);
#pragma unroll 1
    for (int kt = wave; kt < SEQ / 16; kt += 8) {
      const int n0 = kt * 16, n1 = (kt + 4) * 16;
      // phase 0: load tile kt+4, compute tile kt
      v16h b10 = load_bT_f16(Kbh, K_CH, n1 + m, h, 0);
      v16h b11 = load_bT_f16(Kbh, K_CH, n1 + m, h, 32);
      v8f s0 = {};
      s0 = WMMA_F16(aq0, b00, s0);
      s0 = WMMA_F16(aq1, b01, s0);
#pragma unroll
      for (int r = 0; r < 8; ++r)
        Ssc[(size_t)(r + 8 * h) * SEQ + n0 + m] = s0[r] * 0.125f; // 1/sqrt(64)
      // phase 1: load tile kt+8, compute tile kt+4
      if (kt + 8 < SEQ / 16) {
        b00 = load_bT_f16(Kbh, K_CH, (kt + 8) * 16 + m, h, 0);
        b01 = load_bT_f16(Kbh, K_CH, (kt + 8) * 16 + m, h, 32);
      }
      v8f s1 = {};
      s1 = WMMA_F16(aq0, b10, s1);
      s1 = WMMA_F16(aq1, b11, s1);
#pragma unroll
      for (int r = 0; r < 8; ++r)
        Ssc[(size_t)(r + 8 * h) * SEQ + n1 + m] = s1[r] * 0.125f;
    }
  }
  __syncthreads();

  // ---- softmax stats per row of 2048: 8 threads / row ---------------------
  {
    const int r = t >> 3, seg = t & 7;
    float* Srow = Ssc + (size_t)r * SEQ;
    const int c0 = seg * 256;

    float lmax = -3.402823466e38f;
    for (int c = c0; c < c0 + 256; ++c) lmax = fmaxf(lmax, Srow[c]);
    red[r * 8 + seg] = lmax;
    __syncthreads();
    float rmax = red[r * 8];
#pragma unroll
    for (int i = 1; i < 8; ++i) rmax = fmaxf(rmax, red[r * 8 + i]);
    __syncthreads();

    float lsum = 0.f;
    for (int c = c0; c < c0 + 256; ++c) {
      const float e = __expf(Srow[c] - rmax);
      Srow[c] = e;                     // keep exp values; normalize on stream
      lsum += e;
    }
    red[r * 8 + seg] = lsum;
    __syncthreads();
    float rsum = 0.f;
#pragma unroll
    for (int i = 0; i < 8; ++i) rsum += red[r * 8 + i];
    const float inv = 1.0f / rsum;
    if (seg == 0) invRow[r] = inv;

    _Float16* Prow = Pp + (size_t)r * SEQ;
    for (int c = c0; c < c0 + 256; ++c)
      Prow[c] = (_Float16)(Srow[c] * inv);  // f16 probs for P@V
  }
  __syncthreads();

  // ---- stream the 537MB attn matrix out exactly once, non-temporal,
  //      with normalization fused into the store loop ----------------------
  {
    float* dst = attn + ((size_t)bh * SEQ + q0) * SEQ;  // stripe is contiguous
    for (int i = t; i < 16 * SEQ; i += 128)
      __builtin_nontemporal_store(Ssc[i] * invRow[i >> 11], dst + i);
  }

  // ---- P @ V: each wave owns one 16-wide slice of the 64 head dims --------
  {
    const int n0 = wave * 16;
    v8f acc = {};
    v16h pa0 = load_a_f16(Pp, SEQ, m, h, 0);
    v16h vb0 = load_bT_f16(Vbh, SEQ, n0 + m, h, 0);
#pragma unroll 1
    for (int kb = 0; kb < SEQ; kb += 64) {
      // phase 0: load kb+32, consume kb
      v16h pa1 = load_a_f16(Pp, SEQ, m, h, kb + 32);
      v16h vb1 = load_bT_f16(Vbh, SEQ, n0 + m, h, kb + 32);
      acc = WMMA_F16(pa0, vb0, acc);
      // phase 1: load kb+64, consume kb+32
      if (kb + 64 < SEQ) {
        pa0 = load_a_f16(Pp, SEQ, m, h, kb + 64);
        vb0 = load_bT_f16(Vbh, SEQ, n0 + m, h, kb + 64);
      }
      acc = WMMA_F16(pa1, vb1, acc);
    }
    const int bb = bh >> 4, hd = bh & 15;
    _Float16* Ob = O + ((size_t)bb * SEQ + q0) * IN_CH + hd * K_CH + n0 + m;
#pragma unroll
    for (int r = 0; r < 8; ++r)
      Ob[(size_t)(r + 8 * h) * IN_CH] = (_Float16)acc[r];
  }
}

// ---------------------------------------------------------------------------
// Pass 3: output projection  out = O(f16) @ Wo + bo   (fp32 out),
// 16x64 register blocking, same two-phase loop as proj16.
// ---------------------------------------------------------------------------
__global__ void __launch_bounds__(32)
outproj16(const _Float16* __restrict__ A, const _Float16* __restrict__ Wt,
          const float* __restrict__ bias, float* __restrict__ out) {
  const int lane = threadIdx.x & 31;
  const int m = lane & 15, h = lane >> 4;
  const int col0 = blockIdx.x * 64;
  const int row0 = blockIdx.y * 16;
  const _Float16* Arow = A + (size_t)row0 * IN_CH;

  v8f acc0 = {}, acc1 = {}, acc2 = {}, acc3 = {};

  v16h a0  = load_a_f16(Arow, IN_CH, m, h, 0);
  v16h b00 = load_bT_f16(Wt, IN_CH, col0 +  0 + m, h, 0);
  v16h b01 = load_bT_f16(Wt, IN_CH, col0 + 16 + m, h, 0);
  v16h b02 = load_bT_f16(Wt, IN_CH, col0 + 32 + m, h, 0);
  v16h b03 = load_bT_f16(Wt, IN_CH, col0 + 48 + m, h, 0);

#pragma unroll 1
  for (int kb = 0; kb < IN_CH; kb += 64) {
    v16h a1  = load_a_f16(Arow, IN_CH, m, h, kb + 32);
    v16h b10 = load_bT_f16(Wt, IN_CH, col0 +  0 + m, h, kb + 32);
    v16h b11 = load_bT_f16(Wt, IN_CH, col0 + 16 + m, h, kb + 32);
    v16h b12 = load_bT_f16(Wt, IN_CH, col0 + 32 + m, h, kb + 32);
    v16h b13 = load_bT_f16(Wt, IN_CH, col0 + 48 + m, h, kb + 32);
    acc0 = WMMA_F16(a0, b00, acc0);
    acc1 = WMMA_F16(a0, b01, acc1);
    acc2 = WMMA_F16(a0, b02, acc2);
    acc3 = WMMA_F16(a0, b03, acc3);
    if (kb + 64 < IN_CH) {
      a0  = load_a_f16(Arow, IN_CH, m, h, kb + 64);
      b00 = load_bT_f16(Wt, IN_CH, col0 +  0 + m, h, kb + 64);
      b01 = load_bT_f16(Wt, IN_CH, col0 + 16 + m, h, kb + 64);
      b02 = load_bT_f16(Wt, IN_CH, col0 + 32 + m, h, kb + 64);
      b03 = load_bT_f16(Wt, IN_CH, col0 + 48 + m, h, kb + 64);
    }
    acc0 = WMMA_F16(a1, b10, acc0);
    acc1 = WMMA_F16(a1, b11, acc1);
    acc2 = WMMA_F16(a1, b12, acc2);
    acc3 = WMMA_F16(a1, b13, acc3);
  }

  v8f accs[4] = {acc0, acc1, acc2, acc3};
#pragma unroll
  for (int j = 0; j < 4; ++j) {
    const int col = col0 + j * 16 + m;
    const float bv = bias[col];
#pragma unroll
    for (int r = 0; r < 8; ++r)
      out[(size_t)(row0 + r + 8 * h) * IN_CH + col] = accs[j][r] + bv;
  }
}

// ---------------------------------------------------------------------------
extern "C" void kernel_launch(void* const* d_in, const int* in_sizes, int n_in,
                              void* d_out, int out_size, void* d_ws, size_t ws_size,
                              hipStream_t stream) {
  (void)in_sizes; (void)n_in; (void)out_size; (void)ws_size;

  const float* q  = (const float*)d_in[0];
  const float* k  = (const float*)d_in[1];
  const float* v  = (const float*)d_in[2];
  const float* Wq = (const float*)d_in[3];
  const float* bq = (const float*)d_in[4];
  const float* Wk = (const float*)d_in[5];
  const float* bk = (const float*)d_in[6];
  const float* Wv = (const float*)d_in[7];
  const float* bv = (const float*)d_in[8];
  const float* Wo = (const float*)d_in[9];
  const float* bo = (const float*)d_in[10];

  // ---- workspace layout (all f16) ----------------------------------------
  const size_t actElems  = (size_t)ROWS * IN_CH;                 // 4M
  const size_t wElems    = (size_t)IN_CH * IN_CH;                // 1M
  const size_t headElems = (size_t)BATCH * N_HEAD * SEQ * K_CH;  // 4M
  _Float16* Qf  = (_Float16*)d_ws;       // f16 activations
  _Float16* Kf  = Qf  + actElems;
  _Float16* Vf  = Kf  + actElems;
  _Float16* WqT = Vf  + actElems;        // transposed f16 weights [N][K]
  _Float16* WkT = WqT + wElems;
  _Float16* WvT = WkT + wElems;
  _Float16* WoT = WvT + wElems;
  _Float16* Qh  = WoT + wElems;          // [b][h][s][64]
  _Float16* Kh  = Qh  + headElems;       // [b][h][s][64]
  _Float16* Vt  = Kh  + headElems;       // [b][h][64][2048]
  _Float16* O   = Vt  + headElems;       // [b][s][1024]

  float* out  = (float*)d_out;                         // [2,2048,1024]
  float* attn = out + (size_t)BATCH * SEQ * IN_CH;     // [2,16,2048,2048]

  // ---- pass 0: convert activations + transpose/convert weights -----------
  const int nAct = (int)actElems;
  cvt_f16<<<dim3((nAct + 255) / 256), 256, 0, stream>>>(q, Qf, nAct);
  cvt_f16<<<dim3((nAct + 255) / 256), 256, 0, stream>>>(k, Kf, nAct);
  cvt_f16<<<dim3((nAct + 255) / 256), 256, 0, stream>>>(v, Vf, nAct);
  const dim3 wgrid(IN_CH / 32, IN_CH / 32), wblk(32, 8);
  wcvt_t<<<wgrid, wblk, 0, stream>>>(Wq, WqT);
  wcvt_t<<<wgrid, wblk, 0, stream>>>(Wk, WkT);
  wcvt_t<<<wgrid, wblk, 0, stream>>>(Wv, WvT);
  wcvt_t<<<wgrid, wblk, 0, stream>>>(Wo, WoT);

  // ---- pass 1: projections ------------------------------------------------
  const dim3 pgrid(IN_CH / 64, ROWS / 16);
  proj16<<<pgrid, 32, 0, stream>>>(Qf, WqT, bq, Qh, 0);
  proj16<<<pgrid, 32, 0, stream>>>(Kf, WkT, bk, Kh, 0);
  proj16<<<pgrid, 32, 0, stream>>>(Vf, WvT, bv, Vt, 1);

  // ---- pass 2: attention --------------------------------------------------
  const size_t smem = (size_t)16 * SEQ * 4 + (size_t)16 * SEQ * 2
                    + (size_t)16 * 8 * 4 + 16 * 4;
  hipFuncSetAttribute((const void*)attn16,
                      hipFuncAttributeMaxDynamicSharedMemorySize, (int)smem);
  attn16<<<dim3(BATCH * N_HEAD * (SEQ / 16)), 128, smem, stream>>>(Qh, Kh, Vt, attn, O);

  // ---- pass 3: output projection ------------------------------------------
  outproj16<<<dim3(IN_CH / 64, ROWS / 16), 32, 0, stream>>>(O, WoT, bo, out);
}
